// StackedSTU1D_48550310314373
// MI455X (gfx1250) — matrix operand order repaired and checked by reading, not compiled, not run on hardware
//
#include <hip/hip_runtime.h>
#include <hip/hip_bf16.h>
#include <math.h>

// Problem constants (match reference)
#define BB   8
#define CINC 3
#define NN   2048
#define DD   128
#define KKB  16      // K spectral bases
#define LLY  4
#define NRR  1025    // N/2+1
#define NH   1024    // packed complex FFT length

typedef float  v8f   __attribute__((ext_vector_type(8)));
typedef __bf16 v16bf __attribute__((ext_vector_type(16)));

static __device__ __forceinline__ v8f wmma_bf16(v16bf a, v16bf b, v8f c) {
  // D(16x16 f32) = A(16x32 bf16) * B(32x16 bf16) + C
  return __builtin_amdgcn_wmma_f32_16x16x32_bf16(false, a, false, b, (short)0, c,
                                                 false, false);
}

// ---------------------------------------------------------------- weight pack+split
// in: [batch][rows][kdim] f32 (row = output row, kdim = reduction dim)
// out (hi/lo bf16): [batch][c = kdim/32][t = rows/16][lane 0..31][j 0..15]
// in this file's WMMA A-fragment order (contiguous k, same as the B staging):
//   K_in_chunk = 16*(lane>>4) + j
__global__ void k_pack(const float* __restrict__ in, __bf16* __restrict__ hi,
                       __bf16* __restrict__ lo, int batch, int rows, int kdim) {
  int idx = blockIdx.x * blockDim.x + threadIdx.x;
  int perB = rows * kdim;
  if (idx >= batch * perB) return;
  int bt  = idx / perB;
  int rem = idx - bt * perB;
  int j    = rem & 15;
  int lane = (rem >> 4) & 31;
  int r2   = rem >> 9;
  int tcnt = rows >> 4;
  int tt = r2 % tcnt;
  int c  = r2 / tcnt;
  int hs = lane >> 4, m = lane & 15;
  int v = j >> 1, sub = j & 1;
  int Kin = 16 * hs + j;
  int row = tt * 16 + m;
  int col = c * 32 + Kin;
  float x = in[(size_t)bt * perB + (size_t)row * kdim + col];
  __bf16 xh = (__bf16)x;
  hi[idx] = xh;
  lo[idx] = (__bf16)(x - (float)xh);
}

// ---------------------------------------------------------------- lift
__global__ void k_lift(const float* __restrict__ x, const float* __restrict__ lw,
                       const float* __restrict__ lb, float* __restrict__ h) {
  int idx = blockIdx.x * blockDim.x + threadIdx.x;
  if (idx >= BB * DD * NN) return;
  int b   = idx / (DD * NN);
  int rem = idx - b * DD * NN;
  int d   = rem / NN;
  int n   = rem - d * NN;
  float coord = (float)n * (1.0f / (float)(NN - 1));
  const float* xb = x + (size_t)b * CINC * NN + n;
  float acc = lb[d] + lw[d * 4 + 3] * coord;
  acc += lw[d * 4 + 0] * xb[0];
  acc += lw[d * 4 + 1] * xb[NN];
  acc += lw[d * 4 + 2] * xb[2 * NN];
  h[idx] = acc;
}

// ---------------------------------------------------------------- layernorm over D
__global__ void k_ln(const float* __restrict__ h, float* __restrict__ z,
                     const float* __restrict__ g, const float* __restrict__ bta) {
  int idx = blockIdx.x * blockDim.x + threadIdx.x;
  if (idx >= BB * NN) return;
  int b = idx / NN, n = idx - b * NN;
  const float* hp = h + (size_t)b * DD * NN + n;
  float mu = 0.f;
  for (int d = 0; d < DD; ++d) mu += hp[(size_t)d * NN];
  mu *= (1.0f / DD);
  float var = 0.f;
  for (int d = 0; d < DD; ++d) { float v = hp[(size_t)d * NN] - mu; var += v * v; }
  var *= (1.0f / DD);
  float inv = rsqrtf(var + 1e-5f);
  float* zp = z + (size_t)b * DD * NN + n;
  for (int d = 0; d < DD; ++d)
    zp[(size_t)d * NN] = (hp[(size_t)d * NN] - mu) * inv * g[d] + bta[d];
}

// ---------------------------------------------------------------- rfft (2048 real -> 1025 bins)
__global__ void k_rfft(const float* __restrict__ z, float2* __restrict__ Xf) {
  __shared__ float2 A[NH];
  const int row = blockIdx.x;
  const float* src = z + (size_t)row * NN;
  const int t = threadIdx.x;
  for (int m = t; m < NH; m += 256) {
    int rev = (int)(__brev((unsigned)m) >> 22);
    A[rev] = make_float2(src[2 * m], src[2 * m + 1]);
  }
  __syncthreads();
  for (int len = 2; len <= NH; len <<= 1) {
    int half = len >> 1;
    float ab = -6.2831853071795864f / (float)len;
    for (int bf = t; bf < NH / 2; bf += 256) {
      int grp = bf / half;
      int pos = bf - grp * half;
      int i0 = grp * len + pos;
      int i1 = i0 + half;
      float sn, cs;
      __sincosf(ab * (float)pos, &sn, &cs);
      float2 u = A[i0], v = A[i1];
      float2 vw = make_float2(v.x * cs - v.y * sn, v.x * sn + v.y * cs);
      A[i0] = make_float2(u.x + vw.x, u.y + vw.y);
      A[i1] = make_float2(u.x - vw.x, u.y - vw.y);
    }
    __syncthreads();
  }
  float2* dst = Xf + (size_t)row * NRR;
  for (int r = t; r <= NH; r += 256) {
    float2 Yr = A[r & (NH - 1)];
    float2 Ys = A[(NH - r) & (NH - 1)];
    float2 Er = make_float2(0.5f * (Yr.x + Ys.x), 0.5f * (Yr.y - Ys.y));
    float2 Od = make_float2(0.5f * (Yr.y + Ys.y), -0.5f * (Yr.x - Ys.x));
    float sn, cs;
    __sincosf(-6.2831853071795864f * (float)r / (float)NN, &sn, &cs);
    float2 tw = make_float2(Od.x * cs - Od.y * sn, Od.x * sn + Od.y * cs);
    dst[r] = make_float2(Er.x + tw.x, Er.y + tw.y);
  }
}

// ---------------------------------------------------------------- irfft (1025 bins -> 2048 real)
__global__ void k_irfft(const float2* __restrict__ Sf, float* __restrict__ S) {
  __shared__ float2 A[NH];
  const int row = blockIdx.x;
  const float2* src = Sf + (size_t)row * NRR;
  const int t = threadIdx.x;
  for (int m = t; m < NH; m += 256) {
    float2 Xr = src[m];
    float2 Xs = src[NH - m];
    float2 Er = make_float2(0.5f * (Xr.x + Xs.x), 0.5f * (Xr.y - Xs.y));
    float2 Dm = make_float2(0.5f * (Xr.x - Xs.x), 0.5f * (Xr.y + Xs.y));
    float sn, cs;
    __sincosf(6.2831853071795864f * (float)m / (float)NN, &sn, &cs);
    float2 Od = make_float2(Dm.x * cs - Dm.y * sn, Dm.x * sn + Dm.y * cs);
    int rev = (int)(__brev((unsigned)m) >> 22);
    A[rev] = make_float2(Er.x - Od.y, Er.y + Od.x);
  }
  __syncthreads();
  for (int len = 2; len <= NH; len <<= 1) {
    int half = len >> 1;
    float ab = 6.2831853071795864f / (float)len;
    for (int bf = t; bf < NH / 2; bf += 256) {
      int grp = bf / half;
      int pos = bf - grp * half;
      int i0 = grp * len + pos;
      int i1 = i0 + half;
      float sn, cs;
      __sincosf(ab * (float)pos, &sn, &cs);
      float2 u = A[i0], v = A[i1];
      float2 vw = make_float2(v.x * cs - v.y * sn, v.x * sn + v.y * cs);
      A[i0] = make_float2(u.x + vw.x, u.y + vw.y);
      A[i1] = make_float2(u.x - vw.x, u.y - vw.y);
    }
    __syncthreads();
  }
  float* dst = S + (size_t)row * NN;
  const float scl = 1.0f / (float)NH;
  for (int m = t; m < NH; m += 256) {
    float2 y = A[m];
    dst[2 * m]     = y.x * scl;
    dst[2 * m + 1] = y.y * scl;
  }
}

// ---------------------------------------------------------------- spectral GEMM (split-bf16 WMMA)
// Sf[b,h,r](part) = sum_{k,d} Theta[l,k,h,d] * U_part[d,r;k],  U = Xf * conj(Phi)
// grid: (65 r-tiles, 16 = b*2+part). 8 waves/block cover h=0..127.
__global__ void k_spectral(const __bf16* __restrict__ ApkH,  // layer slice [K][4][8][32][16]
                           const __bf16* __restrict__ ApkL,
                           const float2* __restrict__ Xf,
                           const float2* __restrict__ Phi,
                           float* __restrict__ SfF) {
  __shared__ __align__(32) __bf16 Ufh[4 * 512];
  __shared__ __align__(32) __bf16 Ufl[4 * 512];
  const int rt   = blockIdx.x;
  const int bp   = blockIdx.y;
  const int b    = bp >> 1;
  const int part = bp & 1;
  const int r0   = rt * 16;
  const int t    = threadIdx.x;
  const int wave = t >> 5;
  const int lane = t & 31;
  const int m    = lane & 15;
  v8f acc = {0.f, 0.f, 0.f, 0.f, 0.f, 0.f, 0.f, 0.f};

  for (int k = 0; k < KKB; ++k) {
    // stage U fragments (B-operand order): element (c, lane', j) -> K=d = c*32 + (lane'>>4)*16 + j, N = lane'&15
    for (int e = t; e < 4 * 512; e += 256) {
      int j  = e & 15;
      int ln = (e >> 4) & 31;
      int c  = e >> 9;
      int d  = c * 32 + ((ln >> 4) << 4) + j;
      int r  = r0 + (ln & 15);
      float u = 0.f;
      if (r < NRR) {
        float2 xv = Xf[((size_t)b * DD + d) * NRR + r];
        float2 pv = Phi[(size_t)k * NRR + r];
        u = (part == 0) ? (xv.x * pv.x + xv.y * pv.y)
                        : (xv.y * pv.x - xv.x * pv.y);
      }
      __bf16 uh = (__bf16)u;
      Ufh[e] = uh;
      Ufl[e] = (__bf16)(u - (float)uh);
    }
    __syncthreads();
    for (int c = 0; c < 4; ++c) {
      size_t aoff = ((((size_t)k * 4 + c) * 8 + wave) * 32 + lane) * 16;
      v16bf ah = *(const v16bf*)(ApkH + aoff);
      v16bf al = *(const v16bf*)(ApkL + aoff);
      v16bf bh = *(const v16bf*)(Ufh + c * 512 + lane * 16);
      v16bf bl = *(const v16bf*)(Ufl + c * 512 + lane * 16);
      acc = wmma_bf16(ah, bh, acc);
      acc = wmma_bf16(ah, bl, acc);
      acc = wmma_bf16(al, bh, acc);
    }
    __syncthreads();
  }
  const int h0 = wave * 16;
#pragma unroll
  for (int v = 0; v < 8; ++v) {
    int rowv = (lane < 16) ? v : v + 8;
    int r = r0 + m;
    if (r < NRR) {
      SfF[(((size_t)b * DD + h0 + rowv) * NRR + r) * 2 + part] = acc[v];
    }
  }
}

// ---------------------------------------------------------------- fused MLP + residual (split-bf16 WMMA)
// grid: (N/16, B). 8 waves per block.
__global__ void k_mlp(const float* __restrict__ Sg,
                      const __bf16* __restrict__ F1H,  // layer slice [4][16][32][16]
                      const __bf16* __restrict__ F1L,
                      const float* __restrict__ fc1_b,
                      const __bf16* __restrict__ F2H,  // layer slice [8][8][32][16]
                      const __bf16* __restrict__ F2L,
                      const float* __restrict__ fc2_b,
                      float* __restrict__ h) {
  __shared__ __align__(32) __bf16 Sfh[4 * 512];
  __shared__ __align__(32) __bf16 Sfl[4 * 512];
  __shared__ __align__(32) __bf16 H1h[8 * 512];
  __shared__ __align__(32) __bf16 H1l[8 * 512];
  const int n0 = blockIdx.x * 16;
  const int b  = blockIdx.y;
  const int t  = threadIdx.x;
  const int wave = t >> 5;
  const int lane = t & 31;
  const int m    = lane & 15;

  // stage S tile in B-fragment order
  for (int e = t; e < 4 * 512; e += 256) {
    int j  = e & 15;
    int ln = (e >> 4) & 31;
    int c  = e >> 9;
    int d  = c * 32 + ((ln >> 4) << 4) + j;
    float s = Sg[((size_t)b * DD + d) * NN + n0 + (ln & 15)];
    __bf16 sh = (__bf16)s;
    Sfh[e] = sh;
    Sfl[e] = (__bf16)(s - (float)sh);
  }
  __syncthreads();

  // fc1: wave handles row tiles t0=2w, t1=2w+1 (rows hb..hb+31)
  v8f acc0 = {0.f, 0.f, 0.f, 0.f, 0.f, 0.f, 0.f, 0.f};
  v8f acc1 = {0.f, 0.f, 0.f, 0.f, 0.f, 0.f, 0.f, 0.f};
  for (int c = 0; c < 4; ++c) {
    v16bf bh = *(const v16bf*)(Sfh + c * 512 + lane * 16);
    v16bf bl = *(const v16bf*)(Sfl + c * 512 + lane * 16);
    size_t a0 = (((size_t)c * 16 + 2 * wave) * 32 + lane) * 16;
    size_t a1 = (((size_t)c * 16 + 2 * wave + 1) * 32 + lane) * 16;
    v16bf ah0 = *(const v16bf*)(F1H + a0);
    v16bf al0 = *(const v16bf*)(F1L + a0);
    v16bf ah1 = *(const v16bf*)(F1H + a1);
    v16bf al1 = *(const v16bf*)(F1L + a1);
    acc0 = wmma_bf16(ah0, bh, acc0);
    acc0 = wmma_bf16(ah0, bl, acc0);
    acc0 = wmma_bf16(al0, bh, acc0);
    acc1 = wmma_bf16(ah1, bh, acc1);
    acc1 = wmma_bf16(ah1, bl, acc1);
    acc1 = wmma_bf16(al1, bh, acc1);
  }
  // bias + exact gelu, write H1 in B-fragment order (split hi/lo)
#pragma unroll
  for (int v = 0; v < 8; ++v) {
    int rowv = (lane < 16) ? v : v + 8;
#pragma unroll
    for (int tile = 0; tile < 2; ++tile) {
      int hr = (2 * wave + tile) * 16 + rowv;
      float g = (tile == 0 ? acc0[v] : acc1[v]) + fc1_b[hr];
      g = 0.5f * g * (1.0f + erff(g * 0.70710678118654752f));
      __bf16 gh = (__bf16)g;
      int addr = (hr >> 5) * 512 + (((hr >> 4) & 1) * 16 + m) * 16 + (hr & 15);
      H1h[addr] = gh;
      H1l[addr] = (__bf16)(g - (float)gh);
    }
  }
  __syncthreads();

  // fc2: wave handles row tile w (rows 16w..16w+15), K = 256 in 8 chunks
  v8f acc2 = {0.f, 0.f, 0.f, 0.f, 0.f, 0.f, 0.f, 0.f};
  for (int c = 0; c < 8; ++c) {
    v16bf bh = *(const v16bf*)(H1h + c * 512 + lane * 16);
    v16bf bl = *(const v16bf*)(H1l + c * 512 + lane * 16);
    size_t aoff = (((size_t)c * 8 + wave) * 32 + lane) * 16;
    v16bf ah = *(const v16bf*)(F2H + aoff);
    v16bf al = *(const v16bf*)(F2L + aoff);
    acc2 = wmma_bf16(ah, bh, acc2);
    acc2 = wmma_bf16(ah, bl, acc2);
    acc2 = wmma_bf16(al, bh, acc2);
  }
#pragma unroll
  for (int v = 0; v < 8; ++v) {
    int rowv = (lane < 16) ? v : v + 8;
    int d = wave * 16 + rowv;
    size_t off = ((size_t)b * DD + d) * NN + n0 + m;
    h[off] = h[off] + acc2[v] + fc2_b[d];
  }
}

// ---------------------------------------------------------------- head
__global__ void k_head(const float* __restrict__ h, const float* __restrict__ hw,
                       const float* __restrict__ hbias, float* __restrict__ out) {
  int idx = blockIdx.x * blockDim.x + threadIdx.x;
  if (idx >= BB * NN) return;
  int b = idx / NN, n = idx - b * NN;
  const float* hp = h + (size_t)b * DD * NN + n;
  float acc = hbias[0];
  for (int d = 0; d < DD; ++d) acc += hw[d] * hp[(size_t)d * NN];
  out[idx] = acc;
}

// ---------------------------------------------------------------- launch
extern "C" void kernel_launch(void* const* d_in, const int* in_sizes, int n_in,
                              void* d_out, int out_size, void* d_ws, size_t ws_size,
                              hipStream_t stream) {
  const float*  x      = (const float*)d_in[0];
  const float*  lift_w = (const float*)d_in[1];
  const float*  lift_b = (const float*)d_in[2];
  const float*  Theta  = (const float*)d_in[3];
  const float*  ln_g   = (const float*)d_in[4];
  const float*  ln_b   = (const float*)d_in[5];
  const float*  fc1_w  = (const float*)d_in[6];
  const float*  fc1_b  = (const float*)d_in[7];
  const float*  fc2_w  = (const float*)d_in[8];
  const float*  fc2_b  = (const float*)d_in[9];
  const float*  head_w = (const float*)d_in[10];
  const float*  head_b = (const float*)d_in[11];
  const float2* Phi    = (const float2*)d_in[12];

  float*  ws  = (float*)d_ws;
  float*  h   = ws;                              // 2,097,152 f32
  float*  z   = h + (size_t)2097152;             // 2,097,152 f32
  float2* Xf  = (float2*)(z + (size_t)2097152);  // 1,049,600 float2
  float*  SfF = (float*)(Xf + (size_t)1049600);  // 2,099,200 f32
  float*  S   = SfF + (size_t)2099200;           // 2,097,152 f32
  __bf16* ThH = (__bf16*)(S + (size_t)2097152);  // 1,048,576 bf16
  __bf16* ThL = ThH + (size_t)1048576;           // 1,048,576 bf16
  __bf16* F1H = ThL + (size_t)1048576;           //   131,072 bf16
  __bf16* F1L = F1H + (size_t)131072;
  __bf16* F2H = F1L + (size_t)131072;            //   131,072 bf16
  __bf16* F2L = F2H + (size_t)131072;

  // one-time weight split+pack into WMMA A-fragment order (hi/lo bf16)
  k_pack<<<(LLY * KKB * DD * DD) / 256, 256, 0, stream>>>(Theta, ThH, ThL,
                                                          LLY * KKB, DD, DD);
  k_pack<<<(LLY * 2 * DD * DD) / 256, 256, 0, stream>>>(fc1_w, F1H, F1L,
                                                        LLY, 2 * DD, DD);
  k_pack<<<(LLY * DD * 2 * DD) / 256, 256, 0, stream>>>(fc2_w, F2H, F2L,
                                                        LLY, DD, 2 * DD);

  k_lift<<<(BB * DD * NN) / 256, 256, 0, stream>>>(x, lift_w, lift_b, h);

  for (int l = 0; l < LLY; ++l) {
    k_ln<<<(BB * NN) / 256, 256, 0, stream>>>(h, z, ln_g + l * DD, ln_b + l * DD);
    k_rfft<<<BB * DD, 256, 0, stream>>>(z, Xf);
    dim3 gs((NRR + 15) / 16, BB * 2);
    k_spectral<<<gs, 256, 0, stream>>>(ThH + (size_t)l * KKB * DD * DD,
                                       ThL + (size_t)l * KKB * DD * DD,
                                       Xf, Phi, SfF);
    k_irfft<<<BB * DD, 256, 0, stream>>>((const float2*)SfF, S);
    dim3 gm(NN / 16, BB);
    k_mlp<<<gm, 256, 0, stream>>>(S,
                                  F1H + (size_t)l * 2 * DD * DD,
                                  F1L + (size_t)l * 2 * DD * DD,
                                  fc1_b + l * 2 * DD,
                                  F2H + (size_t)l * 2 * DD * DD,
                                  F2L + (size_t)l * 2 * DD * DD,
                                  fc2_b + l * DD, h);
  }

  k_head<<<(BB * NN) / 256, 256, 0, stream>>>(h, head_w, head_b, (float*)d_out);
}
